// RelativeBPEWORDFusionMultiheadAttention_46969762349575
// MI455X (gfx1250) — matrix operand order repaired and checked by reading, not compiled, not run on hardware
//
#include <hip/hip_runtime.h>
#include <stdint.h>

// ---------------------------------------------------------------------------
// RelativeBPEWORDFusionMultiheadAttention for MI455X (gfx1250, wave32, WMMA)
//
// All dense GEMMs -> v_wmma_f32_16x16x32_bf16, fp32 accumulate.
// Every GEMM consumes B as [N][K] ("trans") so both LDS staging and fragment
// loads are fully vectorized (ds_store_b128 / ds_load_b128):
//   - attn_word is stored transposed (aw^T[s][w']) for the F1 GEMM
//   - V projection is stored head-transposed [bH, D, T] for the P@V GEMM
// Pipeline:
//   1) fp32->bf16 conversions
//   2) projections (X @ W^T + b)[*SCALE] -> [bH,T,D] (V -> [bH,D,T])
//   3) all_r = q @ rel^T (N padded 33->48, fp32)
//   4) aw^T = (qw @ kw^T + rel)^T            bf16 [64,512,512]
//   5) F1   = mapping_b @ aw                 bf16 [64,1024,512]
//   6) attn = qb @ kb^T + rel ; += F1 @ M^T  bf16 [64,1024,1024] (L2-resident)
//   7) row softmax (fp32 math)
//   8) ctx = P @ V -> [b,t,E] bf16
//   9) out = ctx @ Wo^T + bo -> fp32 [TB,B,E]
// Workspace: ~325 MB.
// ---------------------------------------------------------------------------

typedef __bf16 bf16;
typedef __bf16 bf16x16 __attribute__((ext_vector_type(16)));
typedef float  f32x8   __attribute__((ext_vector_type(8)));

#define N_E   1024
#define N_H   16
#define N_D   64
#define N_B   4
#define N_TB  1024
#define N_TW  512
#define RELW  48
#define QSCALE 0.125f

#define LDS_STR 72   // (BK=64) + 8 pad, in bf16 elements; 144B rows, 16B aligned

// ---------------------------------------------------------------------------
__global__ __launch_bounds__(256) void cvt_bf16(const float* __restrict__ in,
                                                bf16* __restrict__ out, int n) {
    int i = blockIdx.x * 256 + threadIdx.x;
    if (i < n) out[i] = (bf16)in[i];
}

__global__ __launch_bounds__(256) void relpad_bf16(const float* __restrict__ rel,
                                                   bf16* __restrict__ out) {
    int i = blockIdx.x * 256 + threadIdx.x;       // 48*64
    if (i < RELW * N_D) {
        int r = i >> 6;
        out[i] = (r < 33) ? (bf16)rel[i] : (bf16)0.0f;
    }
}

// ---------------------------------------------------------------------------
// bf16 WMMA GEMM:  C[z] = epilogue( A[z] (M x K) * B[z]^T ),  B global = [N][K]
// block 256 = 8 waves; tile 128(M) x 64(N); BK = 64 (2 WMMA K-steps / stage)
// wave grid 4x2; each wave owns 32x32 = 2x2 fragments.
// OM: 0 bf16 [z][m][n] (+ACC)      1 fp32 [z][m][n]
//     2 bf16 head-split [(z*16+n/64)][T][64]          (z = b)
//     3 bf16 ctx-merge  [(z/16)][m][ (z%16)*64+n ]    (z = b*16+h)
//     4 bf16 head-split transposed [(z*16+n/64)][64][T]  (V projection)
//     5 bf16 transposed [z][n][m]                        (aw^T)
// REL: += allr[z][m][clip(n-m,-16,16)+16]   (allr row stride RELW)
// ---------------------------------------------------------------------------
template<int OM, bool REL, bool ACC, bool BIAS>
__global__ __launch_bounds__(256) void gemm_wmma(
    const bf16*  __restrict__ Ag, size_t a_zstride, int a_zdiv, int lda,
    const bf16*  __restrict__ Bg, size_t b_zstride, int b_zdiv, int ldb,
    const float* __restrict__ bias, float scale,
    const float* __restrict__ allr, size_t allr_zstride,
    void* __restrict__ Cv, size_t c_zstride, int ldc, int Trows,
    int N, int K)
{
    __shared__ bf16 As[128 * LDS_STR];   // [m][k]
    __shared__ bf16 Bn[64 * LDS_STR];    // [n][k]  (k contiguous!)

    const int tid = threadIdx.x;
    const int z   = blockIdx.z;
    const int m0  = blockIdx.y * 128;
    const int n0  = blockIdx.x * 64;

    const bf16* Ab = Ag + (size_t)(z / a_zdiv) * a_zstride;
    const bf16* Bb = Bg + (size_t)(z / b_zdiv) * b_zstride;

    const int wave = tid >> 5;
    const int lane = tid & 31;
    const int wm = wave >> 1;
    const int wn = wave & 1;
    const int lm = lane & 15;
    const int lh = lane >> 4;

    f32x8 acc[2][2];
    #pragma unroll
    for (int i = 0; i < 2; ++i)
        #pragma unroll
        for (int j = 0; j < 2; ++j)
            #pragma unroll
            for (int p = 0; p < 8; ++p) acc[i][j][p] = 0.0f;

    for (int k0 = 0; k0 < K; k0 += 64) {
        // ---- stage A tile: 128 x 64, fully vectorized ----
        #pragma unroll
        for (int it = 0; it < 4; ++it) {
            const int slot = tid + it * 256;
            const int row  = slot >> 3;
            const int kg   = (slot & 7) * 8;
            *(uint4*)&As[row * LDS_STR + kg] =
                *(const uint4*)(Ab + (size_t)(m0 + row) * lda + k0 + kg);
        }
        // ---- stage B tile: 64(n) x 64(k) into Bn[n][k], vectorized ----
        #pragma unroll
        for (int it = 0; it < 2; ++it) {
            const int slot = tid + it * 256;
            const int n    = slot >> 3;
            const int kg   = (slot & 7) * 8;
            uint4 u; u.x = 0u; u.y = 0u; u.z = 0u; u.w = 0u;
            if (n0 + n < N)
                u = *(const uint4*)(Bb + (size_t)(n0 + n) * ldb + k0 + kg);
            *(uint4*)&Bn[n * LDS_STR + kg] = u;
        }
        if (k0 + 64 < K)
            __builtin_prefetch(Ab + (size_t)(m0 + (tid >> 3)) * lda + k0 + 64, 0, 1);

        __syncthreads();

        // ---- 2 WMMA K-steps over the staged 64-wide K slab ----
        #pragma unroll
        for (int kk = 0; kk < 64; kk += 32) {
            // A frag (16x32): lane m=lm, K = lh*16 + e (e0-15)
            bf16x16 af[2];
            #pragma unroll
            for (int i = 0; i < 2; ++i) {
                const bf16* ap = &As[(wm * 32 + i * 16 + lm) * LDS_STR + kk];
                const int kb = lh * 16;
                #pragma unroll
                for (int e = 0; e < 8; ++e) af[i][e]     = ap[kb + e];
                #pragma unroll
                for (int e = 0; e < 8; ++e) af[i][8 + e] = ap[8 + kb + e];
            }
            // B frag (32x16): lane n=lm, K = lh*16 + e  -> 16 contiguous bf16
            bf16x16 bfv[2];
            #pragma unroll
            for (int j = 0; j < 2; ++j) {
                const bf16* bp =
                    &Bn[(wn * 32 + j * 16 + lm) * LDS_STR + kk + lh * 16];
                #pragma unroll
                for (int e = 0; e < 16; ++e) bfv[j][e] = bp[e];
            }
            #pragma unroll
            for (int i = 0; i < 2; ++i)
                #pragma unroll
                for (int j = 0; j < 2; ++j)
                    acc[i][j] = __builtin_amdgcn_wmma_f32_16x16x32_bf16(
                        false, af[i], false, bfv[j], (short)0, acc[i][j],
                        false, false);
        }
        __syncthreads();
    }

    // ---- epilogue ----  D layout: VGPR p -> M = p + 8*lh ; N = lm
    #pragma unroll
    for (int i = 0; i < 2; ++i) {
        #pragma unroll
        for (int j = 0; j < 2; ++j) {
            #pragma unroll
            for (int p = 0; p < 8; ++p) {
                const int mg = m0 + wm * 32 + i * 16 + lh * 8 + p;
                const int ng = n0 + wn * 32 + j * 16 + lm;
                if (ng >= N) continue;
                float v = acc[i][j][p];
                if (BIAS) v += bias[ng];
                v *= scale;
                if (REL) {
                    int r = ng - mg;
                    r = (r < -16) ? -16 : (r > 16 ? 16 : r);
                    v += allr[(size_t)z * allr_zstride + (size_t)mg * RELW + (r + 16)];
                }
                if (OM == 0) {
                    bf16* C = (bf16*)Cv;
                    size_t idx = (size_t)z * c_zstride + (size_t)mg * ldc + ng;
                    if (ACC) v += (float)C[idx];
                    C[idx] = (bf16)v;
                } else if (OM == 1) {
                    float* C = (float*)Cv;
                    C[(size_t)z * c_zstride + (size_t)mg * ldc + ng] = v;
                } else if (OM == 2) {
                    bf16* C = (bf16*)Cv;
                    size_t idx = (((size_t)(z * N_H + (ng >> 6)) * Trows + mg) << 6)
                               + (ng & 63);
                    C[idx] = (bf16)v;
                } else if (OM == 3) {
                    bf16* C = (bf16*)Cv;
                    size_t idx = (((size_t)(z >> 4) * N_TB + mg) << 10)
                               + (size_t)((z & 15) * N_D + ng);
                    C[idx] = (bf16)v;
                } else if (OM == 4) {
                    bf16* C = (bf16*)Cv;
                    size_t idx = ((size_t)(z * N_H + (ng >> 6)) * N_D + (ng & 63))
                               * (size_t)Trows + mg;
                    C[idx] = (bf16)v;
                } else { // OM == 5: transposed [z][n][m]
                    bf16* C = (bf16*)Cv;
                    size_t idx = (size_t)z * c_zstride + (size_t)ng * ldc + mg;
                    C[idx] = (bf16)v;
                }
            }
        }
    }
}

// ---------------------------------------------------------------------------
// row softmax over 1024 bf16 logits, in place (fp32 math)
// ---------------------------------------------------------------------------
__global__ __launch_bounds__(256) void softmax_rows(bf16* __restrict__ attn) {
    __shared__ float red[8];
    bf16* p = attn + (size_t)blockIdx.x * N_TB;
    const int t = threadIdx.x;

    float v[4];
    float mx = -1e30f;
    #pragma unroll
    for (int i = 0; i < 4; ++i) {
        v[i] = (float)p[t + 256 * i];
        mx = fmaxf(mx, v[i]);
    }
    #pragma unroll
    for (int o = 16; o > 0; o >>= 1) mx = fmaxf(mx, __shfl_xor(mx, o, 32));
    if ((t & 31) == 0) red[t >> 5] = mx;
    __syncthreads();
    float m2 = red[0];
    #pragma unroll
    for (int i = 1; i < 8; ++i) m2 = fmaxf(m2, red[i]);
    __syncthreads();

    float s = 0.0f;
    #pragma unroll
    for (int i = 0; i < 4; ++i) { v[i] = __expf(v[i] - m2); s += v[i]; }
    #pragma unroll
    for (int o = 16; o > 0; o >>= 1) s += __shfl_xor(s, o, 32);
    if ((t & 31) == 0) red[t >> 5] = s;
    __syncthreads();
    float s2 = 0.0f;
    #pragma unroll
    for (int i = 0; i < 8; ++i) s2 += red[i];
    const float inv = 1.0f / s2;
    #pragma unroll
    for (int i = 0; i < 4; ++i) p[t + 256 * i] = (bf16)(v[i] * inv);
}

// ---------------------------------------------------------------------------
extern "C" void kernel_launch(void* const* d_in, const int* in_sizes, int n_in,
                              void* d_out, int out_size, void* d_ws, size_t ws_size,
                              hipStream_t stream) {
    (void)in_sizes; (void)n_in; (void)out_size; (void)ws_size;

    const float* query_bpe  = (const float*)d_in[0];   // [1024,4,1024]
    const float* query_word = (const float*)d_in[1];   // [512,4,1024]
    const float* mapping    = (const float*)d_in[2];   // [4,1024,512]
    const float* Wq_bpe  = (const float*)d_in[3];
    const float* bq_bpe  = (const float*)d_in[4];
    const float* Wk_bpe  = (const float*)d_in[5];
    const float* bk_bpe  = (const float*)d_in[6];
    const float* Wq_word = (const float*)d_in[7];
    const float* bq_word = (const float*)d_in[8];
    const float* Wk_word = (const float*)d_in[9];
    const float* bk_word = (const float*)d_in[10];
    const float* Wv = (const float*)d_in[11];
    const float* bv = (const float*)d_in[12];
    const float* Wo = (const float*)d_in[13];
    const float* bo = (const float*)d_in[14];
    const float* relk = (const float*)d_in[15];        // [33,64]

    char* ws = (char*)d_ws;
    size_t off = 0;
    auto alloc = [&](size_t bytes) -> void* {
        void* p = ws + off;
        off += (bytes + 255) & ~(size_t)255;
        return p;
    };

    bf16* qb      = (bf16*)alloc((size_t)N_TB * N_B * N_E * 2);
    bf16* qw      = (bf16*)alloc((size_t)N_TW * N_B * N_E * 2);
    bf16* wqb     = (bf16*)alloc((size_t)N_E * N_E * 2);
    bf16* wkb     = (bf16*)alloc((size_t)N_E * N_E * 2);
    bf16* wqw     = (bf16*)alloc((size_t)N_E * N_E * 2);
    bf16* wkw     = (bf16*)alloc((size_t)N_E * N_E * 2);
    bf16* wv      = (bf16*)alloc((size_t)N_E * N_E * 2);
    bf16* wo      = (bf16*)alloc((size_t)N_E * N_E * 2);
    bf16* mapb    = (bf16*)alloc((size_t)N_B * N_TB * N_TW * 2);
    bf16* relp    = (bf16*)alloc((size_t)RELW * N_D * 2);
    bf16* qbpe_h  = (bf16*)alloc((size_t)64 * N_TB * N_D * 2);
    bf16* kbpe_h  = (bf16*)alloc((size_t)64 * N_TB * N_D * 2);
    bf16* v_hT    = (bf16*)alloc((size_t)64 * N_D * N_TB * 2);   // [bH][D][T]
    bf16* qw_h    = (bf16*)alloc((size_t)64 * N_TW * N_D * 2);
    bf16* kw_h    = (bf16*)alloc((size_t)64 * N_TW * N_D * 2);
    float* allr_b = (float*)alloc((size_t)64 * N_TB * RELW * 4);
    float* allr_w = (float*)alloc((size_t)64 * N_TW * RELW * 4);
    bf16* awT     = (bf16*)alloc((size_t)64 * N_TW * N_TW * 2);  // [bh][s][w']
    bf16* F1      = (bf16*)alloc((size_t)64 * N_TB * N_TW * 2);
    bf16* attn    = (bf16*)alloc((size_t)64 * N_TB * N_TB * 2);
    bf16* ctx     = (bf16*)alloc((size_t)N_B * N_TB * N_E * 2);

    dim3 blk(256);
    auto cvt = [&](const float* src, bf16* dst, int n) {
        cvt_bf16<<<dim3((n + 255) / 256), blk, 0, stream>>>(src, dst, n);
    };

    // --- stage 1: precision conversion ---
    cvt(query_bpe,  qb,   N_TB * N_B * N_E);
    cvt(query_word, qw,   N_TW * N_B * N_E);
    cvt(mapping,    mapb, N_B * N_TB * N_TW);
    cvt(Wq_bpe,  wqb, N_E * N_E);
    cvt(Wk_bpe,  wkb, N_E * N_E);
    cvt(Wq_word, wqw, N_E * N_E);
    cvt(Wk_word, wkw, N_E * N_E);
    cvt(Wv,      wv,  N_E * N_E);
    cvt(Wo,      wo,  N_E * N_E);
    relpad_bf16<<<dim3((RELW * N_D + 255) / 256), blk, 0, stream>>>(relk, relp);

    // --- stage 2: projections ---
    gemm_wmma<2, false, false, true><<<dim3(16, 8, 4), blk, 0, stream>>>(
        qb, 1024, 1, 4096, wqb, 0, 1, 1024, bq_bpe, QSCALE,
        nullptr, 0, qbpe_h, 0, 0, N_TB, 1024, 1024);
    gemm_wmma<2, false, false, true><<<dim3(16, 8, 4), blk, 0, stream>>>(
        qb, 1024, 1, 4096, wkb, 0, 1, 1024, bk_bpe, 1.0f,
        nullptr, 0, kbpe_h, 0, 0, N_TB, 1024, 1024);
    gemm_wmma<4, false, false, true><<<dim3(16, 8, 4), blk, 0, stream>>>(
        qb, 1024, 1, 4096, wv, 0, 1, 1024, bv, 1.0f,
        nullptr, 0, v_hT, 0, 0, N_TB, 1024, 1024);
    gemm_wmma<2, false, false, true><<<dim3(16, 4, 4), blk, 0, stream>>>(
        qw, 1024, 1, 4096, wqw, 0, 1, 1024, bq_word, QSCALE,
        nullptr, 0, qw_h, 0, 0, N_TW, 1024, 1024);
    gemm_wmma<2, false, false, true><<<dim3(16, 4, 4), blk, 0, stream>>>(
        qw, 1024, 1, 4096, wkw, 0, 1, 1024, bk_word, 1.0f,
        nullptr, 0, kw_h, 0, 0, N_TW, 1024, 1024);

    // --- stage 3: all_r = q @ rel^T ---
    gemm_wmma<1, false, false, false><<<dim3(1, 8, 64), blk, 0, stream>>>(
        qbpe_h, (size_t)N_TB * N_D, 1, 64, relp, 0, 1, 64, nullptr, 1.0f,
        nullptr, 0, allr_b, (size_t)N_TB * RELW, RELW, 0, RELW, 64);
    gemm_wmma<1, false, false, false><<<dim3(1, 4, 64), blk, 0, stream>>>(
        qw_h, (size_t)N_TW * N_D, 1, 64, relp, 0, 1, 64, nullptr, 1.0f,
        nullptr, 0, allr_w, (size_t)N_TW * RELW, RELW, 0, RELW, 64);

    // --- stage 4: aw^T = (qw @ kw^T + rel)^T  (store transposed) ---
    gemm_wmma<5, true, false, false><<<dim3(8, 4, 64), blk, 0, stream>>>(
        qw_h, (size_t)N_TW * N_D, 1, 64, kw_h, (size_t)N_TW * N_D, 1, 64,
        nullptr, 1.0f, allr_w, (size_t)N_TW * RELW,
        awT, (size_t)N_TW * N_TW, N_TW, 0, N_TW, 64);

    // --- stage 5: F1[bh] = mapping_b @ aw_bh   (B = awT, trans) ---
    gemm_wmma<0, false, false, false><<<dim3(8, 8, 64), blk, 0, stream>>>(
        mapb, (size_t)N_TB * N_TW, 16, 512, awT, (size_t)N_TW * N_TW, 1, 512,
        nullptr, 1.0f, nullptr, 0,
        F1, (size_t)N_TB * N_TW, N_TW, 0, N_TW, 512);

    // --- stage 6: attn = qb @ kb^T + rel ; attn += F1 @ mapping_b^T ---
    gemm_wmma<0, true, false, false><<<dim3(16, 8, 64), blk, 0, stream>>>(
        qbpe_h, (size_t)N_TB * N_D, 1, 64, kbpe_h, (size_t)N_TB * N_D, 1, 64,
        nullptr, 1.0f, allr_b, (size_t)N_TB * RELW,
        attn, (size_t)N_TB * N_TB, N_TB, 0, N_TB, 64);
    gemm_wmma<0, false, true, false><<<dim3(16, 8, 64), blk, 0, stream>>>(
        F1, (size_t)N_TB * N_TW, 1, 512, mapb, (size_t)N_TB * N_TW, 16, 512,
        nullptr, 1.0f, nullptr, 0,
        attn, (size_t)N_TB * N_TB, N_TB, 0, N_TB, 512);

    // --- stage 7: softmax ---
    softmax_rows<<<dim3(64 * N_TB), blk, 0, stream>>>(attn);

    // --- stage 8: ctx = P @ V  (B = v_hT [bH][D][T], trans) ---
    gemm_wmma<3, false, false, false><<<dim3(1, 8, 64), blk, 0, stream>>>(
        attn, (size_t)N_TB * N_TB, 1, 1024, v_hT, (size_t)N_D * N_TB, 1, 1024,
        nullptr, 1.0f, nullptr, 0,
        ctx, 0, 0, 0, N_D, 1024);

    // --- stage 9: out = ctx @ Wo^T + bo -> fp32 [TB, B, E] ---
    gemm_wmma<1, false, false, true><<<dim3(16, 8, 4), blk, 0, stream>>>(
        ctx, (size_t)N_TB * N_E, 1, 1024, wo, 0, 1, 1024, bo, 1.0f,
        nullptr, 0, d_out, (size_t)N_E, N_B * N_E, 0, N_E, 1024);
}
